// CombinedTransformerRnnDecoderLayer_42803644072182
// MI455X (gfx1250) — matrix operand order repaired and checked by reading, not compiled, not run on hardware
//
#include <hip/hip_runtime.h>
#include <hip/hip_bf16.h>
#include <stdint.h>
#include <stddef.h>

// ---------------------------------------------------------------------------
// Problem constants (from the reference): D=1024, H=16, DFF=4096, B=8, T=S=512
// ---------------------------------------------------------------------------
constexpr int kB = 8, kT = 512, kS = 512, kD = 1024, kH = 16, kDFF = 4096;
constexpr int kM = kB * kT;          // 4096 token rows

#define DEV __device__ __forceinline__

typedef __attribute__((ext_vector_type(16))) __bf16       bf16x16;
typedef __attribute__((ext_vector_type(8)))  float        f32x8;
typedef __attribute__((ext_vector_type(4)))  unsigned int u32x4;

DEV unsigned short f2bf(float f) {           // round-to-nearest-even f32 -> bf16
  union { float f; unsigned u; } x; x.f = f;
  unsigned u = x.u;
  u += 0x7fffu + ((u >> 16) & 1u);
  return (unsigned short)(u >> 16);
}

union Frag {
  bf16x16 v;
  u32x4   q[2];
  unsigned short u[16];
};

DEV f32x8 wmma_bf16(bf16x16 a, bf16x16 b, f32x8 c) {
  return __builtin_amdgcn_wmma_f32_16x16x32_bf16(false, a, false, b, (short)0, c,
                                                 false, false);
}

// A fragment: 16x32 bf16 tile, source row-major with row stride ld (elements).
// Contiguous k order used here: lanes 0-15 rows M=0..15 hold K = 0..15; lanes 16-31 hold
// K = 16..31.
DEV bf16x16 load_a_frag(const unsigned short* A, long ld, int m0, int k0, int lane) {
  const int r  = lane & 15;
  const int hi = lane >> 4;
  const unsigned short* p = A + (long)(m0 + r) * ld + k0 + hi * 16;
  Frag f;
  f.q[0] = *(const u32x4*)(p);
  f.q[1] = *(const u32x4*)(p + 8);
  return f.v;
}

// B fragment (32x16) where B = W^T and W is stored N x K row-major ("TN" GEMM):
// lane n = lane&15, lanes 0-15 carry K=k0..k0+15, lanes 16-31 carry K=k0+16..31.
DEV bf16x16 load_b_fragT(const unsigned short* W, long ld, int n0, int k0, int lane) {
  const int n  = lane & 15;
  const int kk = k0 + ((lane >> 4) << 4);
  const unsigned short* p = W + (long)(n0 + n) * ld + kk;
  Frag f;
  f.q[0] = *(const u32x4*)(p);
  f.q[1] = *(const u32x4*)(p + 8);
  return f.v;
}

// gfx1250 async copy: 16 bytes global -> LDS, tracked by ASYNCcnt
DEV void async_b128(unsigned ldsoff, const void* gaddr) {
  asm volatile("global_load_async_to_lds_b128 %0, %1, off"
               :: "v"(ldsoff), "v"((unsigned long long)(uintptr_t)gaddr)
               : "memory");
}

// ---------------------------------------------------------------------------
// f32 -> bf16 cast (weights, memory bank)
// ---------------------------------------------------------------------------
__global__ void cast_kernel(const float* __restrict__ in,
                            unsigned short* __restrict__ out, long n) {
  long i = (long)blockIdx.x * blockDim.x + threadIdx.x;
  if (i < n) out[i] = f2bf(in[i]);
}

// ---------------------------------------------------------------------------
// LayerNorm over D=1024, one block (256 threads) per row, bf16 output
// ---------------------------------------------------------------------------
__global__ void __launch_bounds__(256)
layernorm_kernel(const float* __restrict__ x, const float* __restrict__ g,
                 const float* __restrict__ b, unsigned short* __restrict__ out) {
  const long row = blockIdx.x;
  const float* xr = x + row * kD;
  __shared__ float red[256];
  float v[4];
  float s = 0.f;
#pragma unroll
  for (int i = 0; i < 4; ++i) { v[i] = xr[threadIdx.x + i * 256]; s += v[i]; }
  red[threadIdx.x] = s; __syncthreads();
  for (int o = 128; o > 0; o >>= 1) {
    if (threadIdx.x < o) red[threadIdx.x] += red[threadIdx.x + o];
    __syncthreads();
  }
  const float mean = red[0] / kD; __syncthreads();
  float sq = 0.f;
#pragma unroll
  for (int i = 0; i < 4; ++i) { float d = v[i] - mean; sq += d * d; }
  red[threadIdx.x] = sq; __syncthreads();
  for (int o = 128; o > 0; o >>= 1) {
    if (threadIdx.x < o) red[threadIdx.x] += red[threadIdx.x + o];
    __syncthreads();
  }
  const float inv = rsqrtf(red[0] / kD + 1e-6f);
#pragma unroll
  for (int i = 0; i < 4; ++i) {
    int c = threadIdx.x + i * 256;
    out[row * kD + c] = f2bf((v[i] - mean) * inv * g[c] + b[c]);
  }
}

// ---------------------------------------------------------------------------
// Y = act((X @ W^T + bias) * scale) (+ residual); X: MxK bf16, W: NxK bf16.
//
// Block = 256 threads (8 waves, 2(M) x 4(N)); block tile 64 x 256.
// Wave tile 32 x 64 = 2x4 WMMA tiles -> 8 v_wmma per K-step from 6 LDS
// fragment loads (A reused 4x across waves, B reused 2x).
// A (64x32) and B (256x32) K-slabs are staged global->LDS with the async
// engine (global_load_async_to_lds_b128), double-buffered and overlapped with
// WMMA via s_wait_asynccnt.
// ---------------------------------------------------------------------------
template <bool RELU>
__global__ void __launch_bounds__(256)
gemm_xwT_kernel(const unsigned short* __restrict__ A,
                const unsigned short* __restrict__ W,
                const float* __restrict__ bias,
                const float* __restrict__ residual,
                unsigned short* __restrict__ out_bf16,
                float* __restrict__ out_f32,
                int M, int N, int K, float scale) {
  __shared__ __align__(16) unsigned short Ash[2][64 * 32];    //  8 KB
  __shared__ __align__(16) unsigned short Bsh[2][256 * 32];   // 32 KB
  const int tid  = threadIdx.x;
  const int lane = tid & 31;
  const int wave = tid >> 5;
  const int wm = wave >> 2;               // 0..1
  const int wn = wave & 3;                // 0..3
  const int m0b = blockIdx.y * 64;
  const int n0b = blockIdx.x * 256;

  // stage K-slab s into buffer buf: 1 async b128 for A + 4 for B per thread
  auto issue = [&](int s, int buf) {
    const int k0 = s * 32;
    {
      const int row = tid >> 2, kk = (tid & 3) * 8;
      async_b128((unsigned)(uintptr_t)&Ash[buf][row * 32 + kk],
                 A + (long)(m0b + row) * K + k0 + kk);
    }
#pragma unroll
    for (int c = 0; c < 4; ++c) {
      const int kk = c * 8;
      async_b128((unsigned)(uintptr_t)&Bsh[buf][tid * 32 + kk],
                 W + (long)(n0b + tid) * K + k0 + kk);
    }
  };

  f32x8 acc[2][4] = {};
  const int nstage = K / 32;
  issue(0, 0);
  for (int s = 0; s < nstage; ++s) {
    const int buf = s & 1;
    if (s + 1 < nstage) {
      issue(s + 1, buf ^ 1);                       // prefetch next slab
      asm volatile("s_wait_asynccnt 5" ::: "memory");   // slab s complete
    } else {
      asm volatile("s_wait_asynccnt 0" ::: "memory");
    }
    __syncthreads();                               // visible to all waves
    bf16x16 af[2], bfr[4];
#pragma unroll
    for (int mt = 0; mt < 2; ++mt)
      af[mt] = load_a_frag(&Ash[buf][0], 32, wm * 32 + mt * 16, 0, lane);
#pragma unroll
    for (int nt = 0; nt < 4; ++nt)
      bfr[nt] = load_b_fragT(&Bsh[buf][0], 32, wn * 64 + nt * 16, 0, lane);
#pragma unroll
    for (int mt = 0; mt < 2; ++mt)
#pragma unroll
      for (int nt = 0; nt < 4; ++nt)
        acc[mt][nt] = wmma_bf16(af[mt], bfr[nt], acc[mt][nt]);
    __syncthreads();                               // slab fully consumed
  }

#pragma unroll
  for (int mt = 0; mt < 2; ++mt) {
#pragma unroll
    for (int nt = 0; nt < 4; ++nt) {
      const int col = n0b + wn * 64 + nt * 16 + (lane & 15);
      const int rb  = m0b + wm * 32 + mt * 16 + ((lane >> 4) << 3);
      const float bv = bias[col];
#pragma unroll
      for (int e = 0; e < 8; ++e) {
        const int row = rb + e;
        float v = (acc[mt][nt][e] + bv) * scale;
        if (RELU) v = v > 0.f ? v : 0.f;
        if (residual) v += residual[(long)row * N + col];
        if (out_f32)  out_f32[(long)row * N + col] = v;
        if (out_bf16) out_bf16[(long)row * N + col] = f2bf(v);
      }
    }
  }
}

// ---------------------------------------------------------------------------
// scores[b,h,q,k] = (q_scaled . k); per-head GEMM M=T, N=S, K=dh=64 (2 WMMAs).
// grid: (S/128, T/16, B*H), block 256 (8 waves along N).
// ---------------------------------------------------------------------------
__global__ void __launch_bounds__(256)
attn_scores_kernel(const unsigned short* __restrict__ Q,
                   const unsigned short* __restrict__ Kmat,
                   float* __restrict__ scores, int causal) {
  const int bh = blockIdx.z, b = bh / kH, h = bh % kH;
  const int lane = threadIdx.x & 31, wave = threadIdx.x >> 5;
  const int m0 = blockIdx.y * 16;
  const int n0 = (blockIdx.x * 8 + wave) * 16;
  const unsigned short* qbase = Q    + (long)b * kT * kD + h * 64;
  const unsigned short* kbase = Kmat + (long)b * kS * kD + h * 64;
  f32x8 acc = {};
#pragma unroll
  for (int k0 = 0; k0 < 64; k0 += 32) {
    bf16x16 a  = load_a_frag (qbase, kD, m0, k0, lane);
    bf16x16 bb = load_b_fragT(kbase, kD, n0, k0, lane);
    acc = wmma_bf16(a, bb, acc);
  }
  float* srow = scores + (long)bh * kT * kS;
  const int col = n0 + (lane & 15);
  const int rb  = m0 + ((lane >> 4) << 3);
#pragma unroll
  for (int e = 0; e < 8; ++e) {
    const int row = rb + e;
    float v = acc[e];
    if (causal && col > row) v = -1e30f;
    srow[(long)row * kS + col] = v;
  }
}

// ---------------------------------------------------------------------------
// Row softmax over S=512 (one block per (b,h,t) row); bf16 probs out and,
// for head 0 of cross-attention, f32 probs into the top_attn output region.
// ---------------------------------------------------------------------------
__global__ void __launch_bounds__(256)
softmax_kernel(const float* __restrict__ scores,
               unsigned short* __restrict__ probs,
               float* __restrict__ top) {
  const long row = blockIdx.x;                  // (b*H + h)*T + t
  const int tid = threadIdx.x;
  const float* s = scores + row * kS;
  const float x0 = s[tid], x1 = s[tid + 256];
  __shared__ float red[256];
  red[tid] = fmaxf(x0, x1); __syncthreads();
  for (int o = 128; o > 0; o >>= 1) {
    if (tid < o) red[tid] = fmaxf(red[tid], red[tid + o]);
    __syncthreads();
  }
  const float m = red[0]; __syncthreads();
  const float e0 = __expf(x0 - m), e1 = __expf(x1 - m);
  red[tid] = e0 + e1; __syncthreads();
  for (int o = 128; o > 0; o >>= 1) {
    if (tid < o) red[tid] += red[tid + o];
    __syncthreads();
  }
  const float inv = 1.0f / red[0];
  const float p0 = e0 * inv, p1 = e1 * inv;
  probs[row * kS + tid]       = f2bf(p0);
  probs[row * kS + tid + 256] = f2bf(p1);
  if (top) {
    const long bh = row / kT, t = row % kT;
    const long h = bh % kH, b = bh / kH;
    if (h == 0) {
      float* tp = top + (b * kT + t) * kS;
      tp[tid] = p0; tp[tid + 256] = p1;
    }
  }
}

// ---------------------------------------------------------------------------
// ctx[b,t,h*64+n] = sum_k probs[b,h,t,k] * v[b,k,h*64+n].
// B operand is K-major -> stage 32x16 V tiles into LDS with the gfx1250
// async engine, then gather the B fragment from LDS.
// grid: (1, T/32, B*H), block 256.
// ---------------------------------------------------------------------------
__global__ void __launch_bounds__(256)
attn_ctx_kernel(const unsigned short* __restrict__ P,
                const unsigned short* __restrict__ V,
                unsigned short* __restrict__ ctx) {
  const int bh = blockIdx.z, b = bh / kH, h = bh % kH;
  const int lane = threadIdx.x & 31, wave = threadIdx.x >> 5;
  const int m0 = blockIdx.y * 32 + (wave >> 2) * 16;
  const int n0 = (wave & 3) * 16;                       // dh = 64 -> 4 N tiles
  __shared__ __align__(16) unsigned short vtile[8][32 * 16];
  const unsigned short* prow  = P + (long)bh * kT * kS;
  const unsigned short* vbase = V + (long)b * kS * kD + h * 64;
  const unsigned ldsoff = (unsigned)(uintptr_t)(&vtile[wave][0]) + (unsigned)(lane * 32);
  f32x8 acc = {};
  for (int k0 = 0; k0 < kS; k0 += 32) {
    // async-stage v[k0+lane][n0..n0+15] (32 bytes/row, one row per lane)
    const unsigned long long ga =
        (unsigned long long)(vbase + (long)(k0 + lane) * kD + n0);
    asm volatile("global_load_async_to_lds_b128 %0, %1, off"
                 :: "v"(ldsoff), "v"(ga) : "memory");
    asm volatile("global_load_async_to_lds_b128 %0, %1, off offset:16"
                 :: "v"(ldsoff), "v"(ga) : "memory");
    asm volatile("s_wait_asynccnt 0" ::: "memory");
    bf16x16 a = load_a_frag(prow, kS, m0, k0, lane);
    Frag fb;
    const int n  = lane & 15;
    const int kb = (lane >> 4) << 4;
#pragma unroll
    for (int i = 0; i < 16; ++i) fb.u[i] = vtile[wave][(kb + i) * 16 + n];
    acc = wmma_bf16(a, fb.v, acc);
  }
  const int col = n0 + (lane & 15);
  const int rb  = m0 + ((lane >> 4) << 3);
#pragma unroll
  for (int e = 0; e < 8; ++e) {
    const int row = rb + e;
    ctx[((long)(b * kT + row)) * kD + h * 64 + col] = f2bf(acc[e]);
  }
}

// ---------------------------------------------------------------------------
// Input-feed LSTM, serial over T.  One 1024-thread workgroup (32 waves);
// x=[e_t|h] (16x2048 bf16), c (8x1024 f32) and gates (8x4096 f32) live in the
// 320 KB WGP LDS pool.  Gates computed with WMMA (M=16 tile covers B=8).
// ---------------------------------------------------------------------------
constexpr unsigned kLstmLds = 16 * 2048 * 2 + 8 * 1024 * 4 + 8 * 4096 * 4; // 224 KB

__global__ void __launch_bounds__(1024)
lstm_kernel(const float* __restrict__ mid,            // (B,T,D)
            const unsigned short* __restrict__ wih,   // (4D,2D) bf16
            const unsigned short* __restrict__ whh,   // (4D, D) bf16
            const float* __restrict__ bih, const float* __restrict__ bhh,
            float* __restrict__ outs) {               // (B,T,D)
  extern __shared__ char smem[];
  unsigned short* xsh = (unsigned short*)smem;                      // 16 x 2048
  float* csh = (float*)(smem + 16 * 2048 * 2);                      // 8 x 1024
  float* gsh = (float*)(smem + 16 * 2048 * 2 + 8 * 1024 * 4);       // 8 x 4096
  const int tid = threadIdx.x;
  const int lane = tid & 31, wave = tid >> 5;
  for (int i = tid; i < 16 * 2048; i += 1024) {                     // zero h + pad
    const int r = i >> 11, c = i & 2047;
    if (r >= 8 || c >= 1024) xsh[i] = 0;
  }
  for (int i = tid; i < 8 * 1024; i += 1024) csh[i] = 0.f;
  __syncthreads();
  for (int t = 0; t < kT; ++t) {
    for (int i = tid; i < 8 * 1024; i += 1024) {                    // stage e_t
      const int b = i >> 10, j = i & 1023;
      xsh[b * 2048 + j] = f2bf(mid[((long)b * kT + t) * kD + j]);
    }
    __syncthreads();
    for (int tile = wave; tile < 256; tile += 32) {                 // 4096 gates
      const int n0 = tile * 16;
      f32x8 acc = {};
      for (int k0 = 0; k0 < 2048; k0 += 32) {                       // x @ wih^T
        bf16x16 a  = load_a_frag (xsh, 2048, 0, k0, lane);
        bf16x16 bb = load_b_fragT(wih, 2048, n0, k0, lane);
        acc = wmma_bf16(a, bb, acc);
      }
      for (int k0 = 0; k0 < 1024; k0 += 32) {                       // h @ whh^T
        bf16x16 a  = load_a_frag (xsh + 1024, 2048, 0, k0, lane);
        bf16x16 bb = load_b_fragT(whh, 1024, n0, k0, lane);
        acc = wmma_bf16(a, bb, acc);
      }
      const int col = n0 + (lane & 15);
      const int rb  = (lane >> 4) << 3;
      const float bsum = bih[col] + bhh[col];
#pragma unroll
      for (int e = 0; e < 8; ++e) {
        const int row = rb + e;
        if (row < 8) gsh[row * 4096 + col] = acc[e] + bsum;
      }
    }
    __syncthreads();
    for (int i = tid; i < 8 * 1024; i += 1024) {                    // cell update
      const int b = i >> 10, j = i & 1023;
      const float ig = gsh[b * 4096 + j];
      const float fg = gsh[b * 4096 + 1024 + j];
      const float gg = gsh[b * 4096 + 2048 + j];
      const float og = gsh[b * 4096 + 3072 + j];
      const float si = 1.f / (1.f + __expf(-ig));
      const float sf = 1.f / (1.f + __expf(-fg));
      const float so = 1.f / (1.f + __expf(-og));
      const float c = sf * csh[i] + si * tanhf(gg);
      csh[i] = c;
      const float hv = so * tanhf(c);
      outs[((long)b * kT + t) * kD + j] = hv;
      xsh[b * 2048 + 1024 + j] = f2bf(hv);                          // feed back
    }
    __syncthreads();
  }
}

__global__ void add_kernel(const float* __restrict__ a, const float* __restrict__ b,
                           float* __restrict__ c, long n) {
  long i = (long)blockIdx.x * blockDim.x + threadIdx.x;
  if (i < n) c[i] = a[i] + b[i];
}

// ---------------------------------------------------------------------------
// Orchestration
// ---------------------------------------------------------------------------
extern "C" void kernel_launch(void* const* d_in, const int* in_sizes, int n_in,
                              void* d_out, int out_size, void* d_ws, size_t ws_size,
                              hipStream_t stream) {
  (void)in_sizes; (void)n_in; (void)out_size; (void)ws_size;
  const float* inputs  = (const float*)d_in[0];
  const float* memory  = (const float*)d_in[1];
  // d_in[2]/d_in[3]: pad masks are all-false in this problem; causality is
  // applied analytically in attn_scores_kernel.
  const float* sa_wq = (const float*)d_in[4];
  const float* sa_wk = (const float*)d_in[5];
  const float* sa_wv = (const float*)d_in[6];
  const float* sa_wo = (const float*)d_in[7];
  const float* ca_wq = (const float*)d_in[8];
  const float* ca_wk = (const float*)d_in[9];
  const float* ca_wv = (const float*)d_in[10];
  const float* ca_wo = (const float*)d_in[11];
  const float* l_wih = (const float*)d_in[12];
  const float* l_whh = (const float*)d_in[13];
  const float* f_w1  = (const float*)d_in[14];
  const float* f_w2  = (const float*)d_in[15];
  const float* ln1_g = (const float*)d_in[16];
  const float* ln2_g = (const float*)d_in[17];
  const float* ffn_g = (const float*)d_in[18];
  const float* ln1_b = (const float*)d_in[19];
  const float* ln2_b = (const float*)d_in[20];
  const float* ffn_b = (const float*)d_in[21];
  const float* sa_bq = (const float*)d_in[22];
  const float* sa_bk = (const float*)d_in[23];
  const float* sa_bv = (const float*)d_in[24];
  const float* sa_bo = (const float*)d_in[25];
  const float* ca_bq = (const float*)d_in[26];
  const float* ca_bk = (const float*)d_in[27];
  const float* ca_bv = (const float*)d_in[28];
  const float* ca_bo = (const float*)d_in[29];
  const float* f_b2  = (const float*)d_in[30];
  const float* l_bih = (const float*)d_in[31];
  const float* l_bhh = (const float*)d_in[32];
  const float* f_b1  = (const float*)d_in[33];

  float* out_f = (float*)d_out;                       // (B,T,D)
  float* top_f = out_f + (size_t)kM * kD;             // (B,T,S)

  size_t off = 0;
  auto carve = [&](size_t bytes) -> char* {
    char* p = (char*)d_ws + off;
    off += (bytes + 255) & ~(size_t)255;
    return p;
  };
  auto a16 = [&](size_t n) { return (unsigned short*)carve(n * 2); };
  auto a32 = [&](size_t n) { return (float*)carve(n * 4); };

  unsigned short *wq_b = a16(kD*kD), *wk_b = a16(kD*kD), *wv_b = a16(kD*kD),
                 *wo_b = a16(kD*kD);
  unsigned short *cwq_b = a16(kD*kD), *cwk_b = a16(kD*kD), *cwv_b = a16(kD*kD),
                 *cwo_b = a16(kD*kD);
  unsigned short *wih_b = a16((size_t)4*kD*2*kD), *whh_b = a16((size_t)4*kD*kD);
  unsigned short *w1_b = a16((size_t)kDFF*kD), *w2_b = a16((size_t)kD*kDFF);
  unsigned short *mem_b = a16((size_t)kM*kD);
  unsigned short *xn_b  = a16((size_t)kM*kD);         // LN output (reused 3x)
  unsigned short *qb = a16((size_t)kM*kD), *kb = a16((size_t)kM*kD),
                 *vb = a16((size_t)kM*kD);
  float *scores = a32((size_t)kB*kH*kT*kS);           // 128 MB, reused
  unsigned short *probs = a16((size_t)kB*kH*kT*kS);   // 64 MB, reused
  unsigned short *ctx_b = a16((size_t)kM*kD);
  float *query_f = a32((size_t)kM*kD);
  float *mid_f   = a32((size_t)kM*kD);                // later reused as xr
  float *lstm_f  = a32((size_t)kM*kD);
  unsigned short *hid_b = a16((size_t)kM*kDFF);

  auto cast = [&](const float* src, unsigned short* dst, long n) {
    cast_kernel<<<dim3((unsigned)((n + 255) / 256)), dim3(256), 0, stream>>>(src, dst, n);
  };
  auto gemm = [&](const unsigned short* A, const unsigned short* W,
                  const float* bias, const float* resid,
                  unsigned short* ob, float* of, int M, int N, int K,
                  float scale, bool relu) {
    dim3 g(N / 256, M / 64);
    if (relu)
      gemm_xwT_kernel<true ><<<g, 256, 0, stream>>>(A, W, bias, resid, ob, of, M, N, K, scale);
    else
      gemm_xwT_kernel<false><<<g, 256, 0, stream>>>(A, W, bias, resid, ob, of, M, N, K, scale);
  };

  // -- one-time-per-call precision staging (weights stay L2-resident) --
  cast(sa_wq, wq_b,  (long)kD*kD);  cast(sa_wk, wk_b,  (long)kD*kD);
  cast(sa_wv, wv_b,  (long)kD*kD);  cast(sa_wo, wo_b,  (long)kD*kD);
  cast(ca_wq, cwq_b, (long)kD*kD);  cast(ca_wk, cwk_b, (long)kD*kD);
  cast(ca_wv, cwv_b, (long)kD*kD);  cast(ca_wo, cwo_b, (long)kD*kD);
  cast(l_wih, wih_b, (long)4*kD*2*kD);
  cast(l_whh, whh_b, (long)4*kD*kD);
  cast(f_w1,  w1_b,  (long)kDFF*kD);
  cast(f_w2,  w2_b,  (long)kD*kDFF);
  cast(memory, mem_b, (long)kM*kD);

  const dim3 blk256(256);
  const dim3 gScore(kS / 128, kT / 16, kB * kH);
  const dim3 gCtx(1, kT / 32, kB * kH);

  // ---- self attention -----------------------------------------------------
  layernorm_kernel<<<kM, blk256, 0, stream>>>(inputs, ln1_g, ln1_b, xn_b);
  gemm(xn_b, wq_b, sa_bq, nullptr, qb, nullptr, kM, kD, kD, 0.125f, false); // 1/sqrt(64)
  gemm(xn_b, wk_b, sa_bk, nullptr, kb, nullptr, kM, kD, kD, 1.f, false);
  gemm(xn_b, wv_b, sa_bv, nullptr, vb, nullptr, kM, kD, kD, 1.f, false);
  attn_scores_kernel<<<gScore, blk256, 0, stream>>>(qb, kb, scores, 1);
  softmax_kernel<<<kB * kH * kT, blk256, 0, stream>>>(scores, probs, nullptr);
  attn_ctx_kernel<<<gCtx, blk256, 0, stream>>>(probs, vb, ctx_b);
  gemm(ctx_b, wo_b, sa_bo, inputs, nullptr, query_f, kM, kD, kD, 1.f, false); // +residual

  // ---- cross attention ----------------------------------------------------
  layernorm_kernel<<<kM, blk256, 0, stream>>>(query_f, ln2_g, ln2_b, xn_b);
  gemm(xn_b, cwq_b, ca_bq, nullptr, qb, nullptr, kM, kD, kD, 0.125f, false);
  gemm(mem_b, cwk_b, ca_bk, nullptr, kb, nullptr, kM, kD, kD, 1.f, false);
  gemm(mem_b, cwv_b, ca_bv, nullptr, vb, nullptr, kM, kD, kD, 1.f, false);
  attn_scores_kernel<<<gScore, blk256, 0, stream>>>(qb, kb, scores, 0);
  softmax_kernel<<<kB * kH * kT, blk256, 0, stream>>>(scores, probs, top_f);
  attn_ctx_kernel<<<gCtx, blk256, 0, stream>>>(probs, vb, ctx_b);
  gemm(ctx_b, cwo_b, ca_bo, nullptr, nullptr, mid_f, kM, kD, kD, 1.f, false);

  // ---- input-feed LSTM (serial) ------------------------------------------
  lstm_kernel<<<1, 1024, kLstmLds, stream>>>(mid_f, wih_b, whh_b, l_bih, l_bhh, lstm_f);
  add_kernel<<<(kM * kD) / 256, blk256, 0, stream>>>(lstm_f, query_f, mid_f,
                                                     (long)kM * kD);  // xr

  // ---- FFN + residual -----------------------------------------------------
  layernorm_kernel<<<kM, blk256, 0, stream>>>(mid_f, ffn_g, ffn_b, xn_b);
  gemm(xn_b, w1_b, f_b1, nullptr, hid_b, nullptr, kM, kDFF, kD, 1.f, true);   // ReLU
  gemm(hid_b, w2_b, f_b2, mid_f, nullptr, out_f, kM, kD, kDFF, 1.f, false);   // +xr
}
